// multi_head_attention_1460288880736
// MI455X (gfx1250) — compile-verified, matrix operand order checked by reading, not run on hardware
//
#include <hip/hip_runtime.h>

// ---------------------------------------------------------------------------
// Causal multi-head attention for MI455X (gfx1250, wave32, WMMA).
// Pipeline: 3x QKV projection GEMMs (f32 -> f16 ws) -> flash attention
// (f16 WMMA, fp32 accumulate/softmax) -> output projection (f16 x f32 -> f32).
// All matmuls use V_WMMA_F32_16X16X32_F16 (fp32 accumulate).
// K-chunk staging uses GLOBAL_LOAD_ASYNC_TO_LDS_B128 when the toolchain
// exposes the builtin (ASYNCcnt path), else falls back to load+ds_store.
// ---------------------------------------------------------------------------

#define D_MODEL  1024
#define N_HEADS  16
#define HEAD_DIM 64
#define SEQ_T    2048
#define BATCH    4
#define M_ROWS   (BATCH * SEQ_T)   // 8192

typedef __attribute__((ext_vector_type(16))) _Float16 v16h;
typedef __attribute__((ext_vector_type(8)))  _Float16 v8h;
typedef __attribute__((ext_vector_type(4)))  _Float16 v4h;
typedef __attribute__((ext_vector_type(8)))  float    v8f;
typedef __attribute__((ext_vector_type(4)))  float    v4f;
typedef __attribute__((ext_vector_type(4)))  int      v4i;

#if defined(__HIP_DEVICE_COMPILE__) && __has_builtin(__builtin_amdgcn_global_load_async_to_lds_b128)
#define HAVE_ASYNC_LDS 1
typedef __attribute__((address_space(1))) v4i gas_v4i;   // global int4
typedef __attribute__((address_space(3))) v4i las_v4i;   // LDS int4
#else
#define HAVE_ASYNC_LDS 0
#endif

__device__ __forceinline__ void async_wait0() {
#if defined(__HIP_DEVICE_COMPILE__) && __has_builtin(__builtin_amdgcn_s_wait_asynccnt)
  __builtin_amdgcn_s_wait_asynccnt(0);
#elif HAVE_ASYNC_LDS
  asm volatile("s_wait_asynccnt 0x0" ::: "memory");
#endif
}

__device__ __forceinline__ v16h cat8(v8h lo, v8h hi) {
  return __builtin_shufflevector(lo, hi, 0,1,2,3,4,5,6,7,8,9,10,11,12,13,14,15);
}

__device__ __forceinline__ v8f wmma_f16(v16h a, v16h b, v8f c) {
  // (neg_a, A, neg_b, B, c_mod, C, reuse_a, reuse_b)
  return __builtin_amdgcn_wmma_f32_16x16x32_f16(false, a, false, b, (short)0, c,
                                                false, false);
}

__device__ __forceinline__ v8f zero8() {
  v8f z;
  #pragma unroll
  for (int j = 0; j < 8; j++) z[j] = 0.f;
  return z;
}

// ---------------------------------------------------------------------------
// GEMM: C[M,N] = A[M,K] @ W[K,N] + bias[N]
// Block tile 128x128, BK=32. 256 threads = 8 waves (2M x 4N); each wave does
// 64x32 = 4x2 WMMA tiles. Register double-buffering: while tile k's WMMAs
// run, tile k+1 is loaded from global into registers, then scattered to LDS
// after the barrier. A staged row-major [m][k] f16; W staged transposed
// [n][k] f16 so all fragment loads are two contiguous 16B LDS reads matching
// the ISA 16-bit fragment layout (k = (i<8?0:16) + (lane<16?0:8) + i%8).
// ---------------------------------------------------------------------------
template <typename AT, typename OT>
__global__ __launch_bounds__(256) void gemm_kernel(
    const AT* __restrict__ A, const float* __restrict__ W,
    const float* __restrict__ bias, OT* __restrict__ C,
    int Mdim, int Ndim, int Kdim)
{
  __shared__ _Float16 As[128][40];   // pad to 40 halves: 16B-aligned rows
  __shared__ _Float16 Bs[128][40];   // [n][k]

  const int tid  = threadIdx.x;
  const int lane = tid & 31;
  const int wave = tid >> 5;
  const int wm   = wave >> 2;       // 0..1  (M group)
  const int wn   = wave & 3;        // 0..3  (N group)
  const int l16  = lane & 15;
  const int half = lane >> 4;       // 0/1 -> +8 k-offset in fragment layout
  const int m0   = blockIdx.y * 128;
  const int n0   = blockIdx.x * 128;

  v8f acc[4][2];
  #pragma unroll
  for (int i = 0; i < 4; i++)
    #pragma unroll
    for (int j = 0; j < 2; j++) acc[i][j] = zero8();

  const int arow = tid >> 3;        // 0..31
  const int acol = (tid & 7) * 4;   // 0..28
  const int brow = tid >> 5;        // 0..7   (k within pass)
  const int bcol = (tid & 31) * 4;  // 0..124 (n)

  v4h areg[4];                      // prefetched A tile (converted to f16)
  v4f wreg[4];                      // prefetched W tile (f32, converted on scatter)

  // ---- preload tile kt=0 into registers ----
  #pragma unroll
  for (int rr = 0; rr < 4; rr++) {
    const AT* ap = A + (size_t)(m0 + arow + rr * 32) * Kdim + acol;
    if constexpr (sizeof(AT) == 4) {
      const v4f av = *(const v4f*)ap;
      areg[rr][0] = (_Float16)av[0]; areg[rr][1] = (_Float16)av[1];
      areg[rr][2] = (_Float16)av[2]; areg[rr][3] = (_Float16)av[3];
    } else {
      areg[rr] = *(const v4h*)ap;
    }
    wreg[rr] = *(const v4f*)(W + (size_t)(brow + rr * 8) * Ndim + n0 + bcol);
  }

  for (int kt = 0; kt < Kdim; kt += 32) {
    // ---- scatter prefetched registers into LDS ----
    #pragma unroll
    for (int rr = 0; rr < 4; rr++) {
      *(v4h*)&As[arow + rr * 32][acol] = areg[rr];
      const int k = brow + rr * 8;
      Bs[bcol + 0][k] = (_Float16)wreg[rr][0];
      Bs[bcol + 1][k] = (_Float16)wreg[rr][1];
      Bs[bcol + 2][k] = (_Float16)wreg[rr][2];
      Bs[bcol + 3][k] = (_Float16)wreg[rr][3];
    }
    __syncthreads();

    // ---- issue next tile's global loads (overlap with WMMAs below) ----
    if (kt + 32 < Kdim) {
      #pragma unroll
      for (int rr = 0; rr < 4; rr++) {
        const AT* ap = A + (size_t)(m0 + arow + rr * 32) * Kdim + kt + 32 + acol;
        if constexpr (sizeof(AT) == 4) {
          const v4f av = *(const v4f*)ap;
          areg[rr][0] = (_Float16)av[0]; areg[rr][1] = (_Float16)av[1];
          areg[rr][2] = (_Float16)av[2]; areg[rr][3] = (_Float16)av[3];
        } else {
          areg[rr] = *(const v4h*)ap;
        }
        wreg[rr] = *(const v4f*)(W + (size_t)(kt + 32 + brow + rr * 8) * Ndim + n0 + bcol);
      }
    }

    // ---- fragment loads + 8 WMMAs ----
    v16h afrag[4];
    #pragma unroll
    for (int mi = 0; mi < 4; mi++) {
      const int r = wm * 64 + mi * 16 + l16;
      v8h lo = *(const v8h*)&As[r][half * 8];
      v8h hi = *(const v8h*)&As[r][16 + half * 8];
      afrag[mi] = cat8(lo, hi);
    }
    v16h bfrag[2];
    #pragma unroll
    for (int ni = 0; ni < 2; ni++) {
      const int r = wn * 32 + ni * 16 + l16;
      v8h lo = *(const v8h*)&Bs[r][half * 8];
      v8h hi = *(const v8h*)&Bs[r][16 + half * 8];
      bfrag[ni] = cat8(lo, hi);
    }
    #pragma unroll
    for (int mi = 0; mi < 4; mi++)
      #pragma unroll
      for (int ni = 0; ni < 2; ni++)
        acc[mi][ni] = wmma_f16(afrag[mi], bfrag[ni], acc[mi][ni]);
    __syncthreads();
  }

  // ---- epilogue: bias + store (C layout: M = j + 8*half, N = l16) ----
  #pragma unroll
  for (int ni = 0; ni < 2; ni++) {
    const int n = n0 + wn * 32 + ni * 16 + l16;
    const float bv = bias[n];
    #pragma unroll
    for (int mi = 0; mi < 4; mi++) {
      #pragma unroll
      for (int j = 0; j < 8; j++) {
        const int m = m0 + wm * 64 + mi * 16 + j + half * 8;
        C[(size_t)m * Ndim + n] = (OT)(acc[mi][ni][j] + bv);
      }
    }
  }
}

// ---------------------------------------------------------------------------
// Flash attention (causal, online softmax).
// Grid: (T/128, B*H). 256 threads = 8 waves; wave w owns 16 q-rows.
// Computes S^T = K·Q^T per 32-key chunk so exp(S^T) in C-layout is *directly*
// a valid B operand for O^T += V^T · P^T (no cross-lane shuffles for P).
// K chunk staged row-major to LDS (async-to-LDS when available) and shared by
// all 8 waves; V staged transposed. Causal gating is wave-uniform, with a
// no-mask fast path for chunks fully below the diagonal.
// ---------------------------------------------------------------------------
__global__ __launch_bounds__(256) void attn_kernel(
    const _Float16* __restrict__ qf, const _Float16* __restrict__ kf,
    const _Float16* __restrict__ vf, _Float16* __restrict__ cf)
{
  __shared__ _Float16 Ks[32][72];   // K chunk row-major [key][d], padded rows
  __shared__ _Float16 Vt[64][40];   // V^T: [d][key], 16B-aligned rows

  const int tid  = threadIdx.x;
  const int lane = tid & 31;
  const int wave = tid >> 5;
  const int l16  = lane & 15;
  const int half = lane >> 4;
  const int qblock = blockIdx.x;
  const int bh = blockIdx.y;
  const int b = bh / N_HEADS;
  const int h = bh % N_HEADS;

  const int q0    = qblock * 128 + wave * 16;
  const int q_abs = q0 + l16;                       // this lane's query column
  const size_t base = ((size_t)b * SEQ_T) * D_MODEL + (size_t)h * HEAD_DIM;

  // Q^T fragments (B operand: n = q = l16, elements i: hd = (i<8?0:16) + 8*half + i%8) — held.
  v16h qfrag[2];
  #pragma unroll
  for (int ds = 0; ds < 2; ds++) {
    const _Float16* qr = qf + base + (size_t)q_abs * D_MODEL + ds * 32;
    v8h lo = *(const v8h*)(qr + half * 8);
    v8h hi = *(const v8h*)(qr + 16 + half * 8);
    qfrag[ds] = cat8(lo, hi);
  }

  v8f acc[4];                       // O^T accumulators: 4 d-tiles of 16x16
  #pragma unroll
  for (int dt = 0; dt < 4; dt++) acc[dt] = zero8();
  float m_run = -1e30f, l_run = 0.f;
  const float scale = 0.125f;       // 1/sqrt(64)

  const int kmax = qblock * 128 + 128;   // block-causal key bound
  const int wave_qmax = q0 + 15;
  const int skey = tid >> 3;             // 0..31
  const int sd8  = (tid & 7) * 8;        // 0..56

  for (int c = 0; c < kmax; c += 32) {
    // ---- stage K chunk row-major (async DMA to LDS when available) ----
    {
      const _Float16* kg = kf + base + (size_t)(c + skey) * D_MODEL + sd8;
      _Float16* kl = &Ks[skey][sd8];
#if HAVE_ASYNC_LDS
      __builtin_amdgcn_global_load_async_to_lds_b128((gas_v4i*)kg, (las_v4i*)kl, 0, 0);
#else
      *(v8h*)kl = *(const v8h*)kg;
#endif
    }
    // ---- stage V^T (manual 16-bit transpose) ----
    {
      v8h vv = *(const v8h*)(vf + base + (size_t)(c + skey) * D_MODEL + sd8);
      #pragma unroll
      for (int i = 0; i < 8; i++) Vt[sd8 + i][skey] = vv[i];
    }
    async_wait0();
    __syncthreads();

    if (c <= wave_qmax) {           // wave-uniform causal chunk gate
      // ---- load all K fragments from LDS first, then WMMA chain ----
      v16h kfr[2][2];
      #pragma unroll
      for (int kt2 = 0; kt2 < 2; kt2++) {
        const _Float16* krow = &Ks[kt2 * 16 + l16][0];
        #pragma unroll
        for (int ds = 0; ds < 2; ds++) {
          v8h lo = *(const v8h*)(krow + ds * 32 + half * 8);
          v8h hi = *(const v8h*)(krow + ds * 32 + 16 + half * 8);
          kfr[kt2][ds] = cat8(lo, hi);
        }
      }
      v8f s[2];
      #pragma unroll
      for (int kt2 = 0; kt2 < 2; kt2++) {
        s[kt2] = zero8();
        s[kt2] = wmma_f16(kfr[kt2][0], qfrag[0], s[kt2]);
        s[kt2] = wmma_f16(kfr[kt2][1], qfrag[1], s[kt2]);
      }
      // ---- scale (+ causal mask only near the diagonal); chunk max ----
      float sv[2][8];
      float cmax = -1e30f;
      if (c + 31 <= q0) {
        // fast path: chunk fully below the diagonal for every lane
        #pragma unroll
        for (int kt2 = 0; kt2 < 2; kt2++)
          #pragma unroll
          for (int j = 0; j < 8; j++) {
            const float v = s[kt2][j] * scale;
            sv[kt2][j] = v;
            cmax = fmaxf(cmax, v);
          }
      } else {
        #pragma unroll
        for (int kt2 = 0; kt2 < 2; kt2++)
          #pragma unroll
          for (int j = 0; j < 8; j++) {
            const int key = c + kt2 * 16 + j + half * 8;
            float v = s[kt2][j] * scale;
            v = (key <= q_abs) ? v : -1e30f;
            sv[kt2][j] = v;
            cmax = fmaxf(cmax, v);
          }
      }
      cmax = fmaxf(cmax, __shfl_xor(cmax, 16));
      const float mnew = fmaxf(m_run, cmax);
      const float corr = __expf(m_run - mnew);
      m_run = mnew;
      // ---- P^T = exp(S^T - m): lands directly in B-fragment layout ----
      float rsum = 0.f;
      v16h pfrag;
      #pragma unroll
      for (int j = 0; j < 8; j++) {
        const float p0 = __expf(sv[0][j] - m_run);
        const float p1 = __expf(sv[1][j] - m_run);
        rsum += p0 + p1;
        pfrag[j]     = (_Float16)p0;
        pfrag[8 + j] = (_Float16)p1;
      }
      rsum += __shfl_xor(rsum, 16);
      l_run = l_run * corr + rsum;
      // ---- rescale + O^T += V^T · P^T ----
      #pragma unroll
      for (int dt = 0; dt < 4; dt++) {
        #pragma unroll
        for (int j = 0; j < 8; j++) acc[dt][j] *= corr;
        const int d = dt * 16 + l16;
        v8h lo = *(const v8h*)&Vt[d][half * 8];
        v8h hi = *(const v8h*)&Vt[d][16 + half * 8];
        acc[dt] = wmma_f16(cat8(lo, hi), pfrag, acc[dt]);
      }
    }
    __syncthreads();
  }

  // ---- normalize + write context (f16): lane owns one q row ----
  const float inv = 1.f / l_run;
  _Float16* crow = cf + base + (size_t)q_abs * D_MODEL;
  #pragma unroll
  for (int dt = 0; dt < 4; dt++) {
    v8h o;
    #pragma unroll
    for (int j = 0; j < 8; j++) o[j] = (_Float16)(acc[dt][j] * inv);
    *(v8h*)(crow + dt * 16 + half * 8) = o;   // d = dt*16 + 8*half + j
  }
}

// ---------------------------------------------------------------------------
extern "C" void kernel_launch(void* const* d_in, const int* in_sizes, int n_in,
                              void* d_out, int out_size, void* d_ws, size_t ws_size,
                              hipStream_t stream) {
  const float* q   = (const float*)d_in[0];
  const float* k   = (const float*)d_in[1];
  const float* v   = (const float*)d_in[2];
  const float* w_q = (const float*)d_in[3];
  const float* b_q = (const float*)d_in[4];
  const float* w_k = (const float*)d_in[5];
  const float* b_k = (const float*)d_in[6];
  const float* w_v = (const float*)d_in[7];
  const float* b_v = (const float*)d_in[8];
  const float* w_o = (const float*)d_in[9];
  const float* b_o = (const float*)d_in[10];

  // workspace: 4 x (8192*1024) f16 = 64 MB
  _Float16* qf = (_Float16*)d_ws;
  _Float16* kf = qf + (size_t)M_ROWS * D_MODEL;
  _Float16* vf = kf + (size_t)M_ROWS * D_MODEL;
  _Float16* cf = vf + (size_t)M_ROWS * D_MODEL;

  const dim3 gg(D_MODEL / 128, M_ROWS / 128);   // (8, 64)
  gemm_kernel<float, _Float16><<<gg, 256, 0, stream>>>(q, w_q, b_q, qf, M_ROWS, D_MODEL, D_MODEL);
  gemm_kernel<float, _Float16><<<gg, 256, 0, stream>>>(k, w_k, b_k, kf, M_ROWS, D_MODEL, D_MODEL);
  gemm_kernel<float, _Float16><<<gg, 256, 0, stream>>>(v, w_v, b_v, vf, M_ROWS, D_MODEL, D_MODEL);

  attn_kernel<<<dim3(SEQ_T / 128, BATCH * N_HEADS), 256, 0, stream>>>(qf, kf, vf, cf);

  gemm_kernel<_Float16, float><<<gg, 256, 0, stream>>>(cf, w_o, b_o, (float*)d_out,
                                                       M_ROWS, D_MODEL, D_MODEL);
}
